// BmmMoeModel_36567351558811
// MI455X (gfx1250) — matrix operand order repaired and checked by reading, not compiled, not run on hardware
//
#include <hip/hip_runtime.h>

// Problem constants (B=2, S=2048, H=2048, I=1024, E=8)
#define T_TOK 4096
#define H_DIM 2048
#define I_DIM 1024
#define E_NUM 8

// GEMM tiling
#define BM 64
#define BN 128
#define KB 32

typedef __attribute__((ext_vector_type(16))) __bf16 v16bf;
typedef __attribute__((ext_vector_type(8)))  float  v8f;
typedef __attribute__((ext_vector_type(2)))  float  f32x2;
typedef __attribute__((ext_vector_type(2)))  __bf16 bf16x2;

union Frag32 { unsigned int u[8]; v16bf v; };

// two fp32 -> packed bf16 dword: single v_cvt_pk_bf16_f32
__device__ __forceinline__ unsigned int pack2bf(float a, float b) {
    f32x2 f; f[0] = a; f[1] = b;
    union { bf16x2 h; unsigned int u; } p;
    p.h = __builtin_convertvector(f, bf16x2);
    return p.u;
}

__device__ __forceinline__ v8f wmma_bf16(v16bf a, v16bf b, v8f c) {
    return __builtin_amdgcn_wmma_f32_16x16x32_bf16(
        false, a, false, b, (short)0, c, false, false);
}

__device__ __forceinline__ v8f v8f_zero() {
    v8f z;
#pragma unroll
    for (int j = 0; j < 8; ++j) z[j] = 0.f;
    return z;
}

// ---------------------------------------------------------------------------
// Kernel 0: zero expert counters
// ---------------------------------------------------------------------------
__global__ __launch_bounds__(32) void moe_zero(int* __restrict__ counts) {
    if (threadIdx.x < 16) counts[threadIdx.x] = 0;
}

// ---------------------------------------------------------------------------
// Kernel 1: router — one wave per token (wave32), argmax + sigmoid scale,
//           build per-expert token lists.
// ---------------------------------------------------------------------------
__global__ __launch_bounds__(32) void moe_router(
    const float* __restrict__ x, const float* __restrict__ gw,
    float* __restrict__ scale, int* __restrict__ counts, int* __restrict__ tlist)
{
    const int t = blockIdx.x;
    const int lane = threadIdx.x;
    const float* xr = x + (size_t)t * H_DIM;

    float acc[E_NUM];
#pragma unroll
    for (int e = 0; e < E_NUM; ++e) acc[e] = 0.f;

    for (int k = lane; k < H_DIM; k += 32) {
        const float xv = xr[k];
#pragma unroll
        for (int e = 0; e < E_NUM; ++e) acc[e] += xv * gw[e * H_DIM + k];
    }
#pragma unroll
    for (int e = 0; e < E_NUM; ++e) {
        float v = acc[e];
#pragma unroll
        for (int off = 16; off > 0; off >>= 1) v += __shfl_xor(v, off, 32);
        acc[e] = v;
    }
    if (lane == 0) {
        int be = 0; float bv = acc[0];
#pragma unroll
        for (int e = 1; e < E_NUM; ++e)
            if (acc[e] > bv) { bv = acc[e]; be = e; }   // first-max tie break
        scale[t] = 1.f / (1.f + __expf(-bv));
        const int pos = atomicAdd(&counts[be], 1);
        tlist[be * T_TOK + pos] = t;
    }
}

// ---------------------------------------------------------------------------
// Kernel 2: gathered GEMM1 (x -> gate_up) fused with up*silu(gate).
//           Double-buffered LDS, packed bf16 staging, bf16 WMMA.
// ---------------------------------------------------------------------------
__global__ __launch_bounds__(256) void moe_gemm1(
    const float* __restrict__ x, const float* __restrict__ gup,
    const float* __restrict__ scale, const int* __restrict__ counts,
    const int* __restrict__ tlist, __bf16* __restrict__ act)
{
    const int e  = blockIdx.z;
    const int mt = blockIdx.y;
    const int nb = blockIdx.x;
    const int cnt = counts[e];
    if (mt * BM >= cnt) return;

    __shared__ unsigned short sA [2][BM][KB];
    __shared__ unsigned short sBu[2][BN][KB];
    __shared__ unsigned short sBg[2][BN][KB];

    const int tid  = threadIdx.x;
    const int lane = tid & 31;
    const int wave = tid >> 5;
    const int mw = wave & 3;    // 16-row slice
    const int nw = wave >> 2;   // 64-col slice

    // ---- A staging: thread fills 8 bf16 of one row (one b128 store)
    const int arow = tid >> 2;
    const int aseg = (tid & 3) * 8;
    const int gr = mt * BM + arow;
    int atok = 0; float asc = 0.f;
    if (gr < cnt) { atok = tlist[e * T_TOK + gr]; asc = scale[atok]; }
    const float* xrow = x + (size_t)atok * H_DIM + aseg;

    // ---- B staging: thread covers 8 cols x one k-pair (packed b32 stores)
    const int bcol = (tid & 15) * 8;
    const int bkp  = tid >> 4;                 // k = 2*bkp, 2*bkp+1
    const float* gb = gup + (size_t)e * H_DIM * (2 * I_DIM) + nb * BN + bcol;

    v8f cU[4], cG[4];
#pragma unroll
    for (int i = 0; i < 4; ++i) { cU[i] = v8f_zero(); cG[i] = v8f_zero(); }

    float4 aR0, aR1, u00, u01, u10, u11, g00, g01, g10, g11;

    auto load_global = [&](int k0) {
        aR0 = *reinterpret_cast<const float4*>(xrow + k0);
        aR1 = *reinterpret_cast<const float4*>(xrow + k0 + 4);
        const float* r0 = gb + (size_t)(k0 + 2 * bkp) * (2 * I_DIM);
        const float* r1 = r0 + (2 * I_DIM);
        u00 = *reinterpret_cast<const float4*>(r0);
        u01 = *reinterpret_cast<const float4*>(r0 + 4);
        u10 = *reinterpret_cast<const float4*>(r1);
        u11 = *reinterpret_cast<const float4*>(r1 + 4);
        g00 = *reinterpret_cast<const float4*>(r0 + I_DIM);
        g01 = *reinterpret_cast<const float4*>(r0 + I_DIM + 4);
        g10 = *reinterpret_cast<const float4*>(r1 + I_DIM);
        g11 = *reinterpret_cast<const float4*>(r1 + I_DIM + 4);
    };

    auto store_lds = [&](int buf) {
        uint4 ap;
        ap.x = pack2bf(asc * aR0.x, asc * aR0.y);
        ap.y = pack2bf(asc * aR0.z, asc * aR0.w);
        ap.z = pack2bf(asc * aR1.x, asc * aR1.y);
        ap.w = pack2bf(asc * aR1.z, asc * aR1.w);
        *reinterpret_cast<uint4*>(&sA[buf][arow][aseg]) = ap;
        unsigned short* bu = &sBu[buf][bcol][2 * bkp];
        unsigned short* bg = &sBg[buf][bcol][2 * bkp];
        *reinterpret_cast<unsigned int*>(bu + 0 * KB) = pack2bf(u00.x, u10.x);
        *reinterpret_cast<unsigned int*>(bu + 1 * KB) = pack2bf(u00.y, u10.y);
        *reinterpret_cast<unsigned int*>(bu + 2 * KB) = pack2bf(u00.z, u10.z);
        *reinterpret_cast<unsigned int*>(bu + 3 * KB) = pack2bf(u00.w, u10.w);
        *reinterpret_cast<unsigned int*>(bu + 4 * KB) = pack2bf(u01.x, u11.x);
        *reinterpret_cast<unsigned int*>(bu + 5 * KB) = pack2bf(u01.y, u11.y);
        *reinterpret_cast<unsigned int*>(bu + 6 * KB) = pack2bf(u01.z, u11.z);
        *reinterpret_cast<unsigned int*>(bu + 7 * KB) = pack2bf(u01.w, u11.w);
        *reinterpret_cast<unsigned int*>(bg + 0 * KB) = pack2bf(g00.x, g10.x);
        *reinterpret_cast<unsigned int*>(bg + 1 * KB) = pack2bf(g00.y, g10.y);
        *reinterpret_cast<unsigned int*>(bg + 2 * KB) = pack2bf(g00.z, g10.z);
        *reinterpret_cast<unsigned int*>(bg + 3 * KB) = pack2bf(g00.w, g10.w);
        *reinterpret_cast<unsigned int*>(bg + 4 * KB) = pack2bf(g01.x, g11.x);
        *reinterpret_cast<unsigned int*>(bg + 5 * KB) = pack2bf(g01.y, g11.y);
        *reinterpret_cast<unsigned int*>(bg + 6 * KB) = pack2bf(g01.z, g11.z);
        *reinterpret_cast<unsigned int*>(bg + 7 * KB) = pack2bf(g01.w, g11.w);
    };

    auto compute = [&](int buf) {
        const int r  = mw * 16 + (lane & 15);
        const int kb = (lane < 16) ? 0 : 16;
        Frag32 af;
        const uint4 lo = *reinterpret_cast<const uint4*>(&sA[buf][r][kb]);
        const uint4 hi = *reinterpret_cast<const uint4*>(&sA[buf][r][kb + 8]);
        af.u[0] = lo.x; af.u[1] = lo.y; af.u[2] = lo.z; af.u[3] = lo.w;
        af.u[4] = hi.x; af.u[5] = hi.y; af.u[6] = hi.z; af.u[7] = hi.w;
        const int ksel = (lane < 16) ? 0 : 16;
        v16bf bfr[8];
#pragma unroll
        for (int nt = 0; nt < 4; ++nt) {
            const int cb = nw * 64 + nt * 16 + (lane & 15);
            bfr[nt]     = *reinterpret_cast<const v16bf*>(&sBu[buf][cb][ksel]);
            bfr[4 + nt] = *reinterpret_cast<const v16bf*>(&sBg[buf][cb][ksel]);
        }
#pragma unroll
        for (int nt = 0; nt < 4; ++nt) cU[nt] = wmma_bf16(af.v, bfr[nt], cU[nt]);
#pragma unroll
        for (int nt = 0; nt < 4; ++nt) cG[nt] = wmma_bf16(af.v, bfr[4 + nt], cG[nt]);
    };

    load_global(0);
    store_lds(0);
    int cur = 0;
    for (int k0 = 0; k0 < H_DIM; k0 += KB) {
        __syncthreads();
        const bool more = (k0 + KB) < H_DIM;
        if (more) load_global(k0 + KB);
        compute(cur);
        if (more) store_lds(cur ^ 1);
        cur ^= 1;
    }

    // ---- fused epilogue: act = up * silu(gate), scatter by token id (bf16)
    const int hi8 = (lane < 16) ? 0 : 8;
    const int rowbase = mt * BM + mw * 16 + hi8;
    int tokv[8];
#pragma unroll
    for (int j = 0; j < 8; ++j) {
        const int g2 = rowbase + j;
        tokv[j] = (g2 < cnt) ? tlist[e * T_TOK + g2] : -1;
    }
#pragma unroll
    for (int nt = 0; nt < 4; ++nt) {
        const int col = nb * BN + nw * 64 + nt * 16 + (lane & 15);
#pragma unroll
        for (int j = 0; j < 8; ++j) {
            if (tokv[j] >= 0) {
                const float g = cG[nt][j];
                const float a = cU[nt][j] * (g / (1.f + __expf(-g)));
                act[(size_t)tokv[j] * I_DIM + col] = (__bf16)a;
            }
        }
    }
}

// ---------------------------------------------------------------------------
// Kernel 3: gathered GEMM2 (act -> down), fp32 scatter to output.
//           A operand staged with GLOBAL_LOAD_ASYNC_TO_LDS_B128 (ASYNCcnt).
// ---------------------------------------------------------------------------
__global__ __launch_bounds__(256) void moe_gemm2(
    const __bf16* __restrict__ act, const float* __restrict__ dw,
    const int* __restrict__ counts, const int* __restrict__ tlist,
    float* __restrict__ out)
{
    const int e  = blockIdx.z;
    const int mt = blockIdx.y;
    const int nb = blockIdx.x;
    const int cnt = counts[e];
    if (mt * BM >= cnt) return;

    __shared__ unsigned short sA[2][BM][KB];
    __shared__ unsigned short sB[2][BN][KB];

    const int tid  = threadIdx.x;
    const int lane = tid & 31;
    const int wave = tid >> 5;
    const int mw = wave & 3;
    const int nw = wave >> 2;

    // ---- A: async 16B copy per thread (act rows are already bf16).
    //      Rows past cnt clamp to a valid row; they only feed unsaved outputs.
    const int arow = tid >> 2;
    const int aseg = (tid & 3) * 8;
    const int gr = mt * BM + arow;
    const int grc = (gr < cnt) ? gr : (cnt - 1);
    const int atok = tlist[e * T_TOK + grc];
    const __bf16* arow_p = act + (size_t)atok * I_DIM + aseg;

    auto a_async = [&](int k0, int buf) {
        const unsigned lds_off =
            (unsigned)(size_t)(&sA[buf][arow][aseg]);      // generic low 32b = LDS offset
        const unsigned long long ga =
            (unsigned long long)(size_t)(arow_p + k0);
        asm volatile("global_load_async_to_lds_b128 %0, %1, off"
                     :: "v"(lds_off), "v"(ga) : "memory");
    };

    // ---- B staging: 8 cols x one k-pair, packed b32 stores
    const int bcol = (tid & 15) * 8;
    const int bkp  = tid >> 4;
    const float* db = dw + (size_t)e * I_DIM * H_DIM + nb * BN + bcol;

    float4 b00, b01, b10, b11;
    auto loadB = [&](int k0) {
        const float* r0 = db + (size_t)(k0 + 2 * bkp) * H_DIM;
        const float* r1 = r0 + H_DIM;
        b00 = *reinterpret_cast<const float4*>(r0);
        b01 = *reinterpret_cast<const float4*>(r0 + 4);
        b10 = *reinterpret_cast<const float4*>(r1);
        b11 = *reinterpret_cast<const float4*>(r1 + 4);
    };
    auto storeB = [&](int buf) {
        unsigned short* bp = &sB[buf][bcol][2 * bkp];
        *reinterpret_cast<unsigned int*>(bp + 0 * KB) = pack2bf(b00.x, b10.x);
        *reinterpret_cast<unsigned int*>(bp + 1 * KB) = pack2bf(b00.y, b10.y);
        *reinterpret_cast<unsigned int*>(bp + 2 * KB) = pack2bf(b00.z, b10.z);
        *reinterpret_cast<unsigned int*>(bp + 3 * KB) = pack2bf(b00.w, b10.w);
        *reinterpret_cast<unsigned int*>(bp + 4 * KB) = pack2bf(b01.x, b11.x);
        *reinterpret_cast<unsigned int*>(bp + 5 * KB) = pack2bf(b01.y, b11.y);
        *reinterpret_cast<unsigned int*>(bp + 6 * KB) = pack2bf(b01.z, b11.z);
        *reinterpret_cast<unsigned int*>(bp + 7 * KB) = pack2bf(b01.w, b11.w);
    };

    v8f cc[4];
#pragma unroll
    for (int i = 0; i < 4; ++i) cc[i] = v8f_zero();

    auto compute = [&](int buf) {
        const int r  = mw * 16 + (lane & 15);
        const int kb = (lane < 16) ? 0 : 16;
        Frag32 af;
        const uint4 lo = *reinterpret_cast<const uint4*>(&sA[buf][r][kb]);
        const uint4 hi = *reinterpret_cast<const uint4*>(&sA[buf][r][kb + 8]);
        af.u[0] = lo.x; af.u[1] = lo.y; af.u[2] = lo.z; af.u[3] = lo.w;
        af.u[4] = hi.x; af.u[5] = hi.y; af.u[6] = hi.z; af.u[7] = hi.w;
        const int ksel = (lane < 16) ? 0 : 16;
        v16bf bfr[4];
#pragma unroll
        for (int nt = 0; nt < 4; ++nt) {
            const int cb = nw * 64 + nt * 16 + (lane & 15);
            bfr[nt] = *reinterpret_cast<const v16bf*>(&sB[buf][cb][ksel]);
        }
#pragma unroll
        for (int nt = 0; nt < 4; ++nt) cc[nt] = wmma_bf16(af.v, bfr[nt], cc[nt]);
    };

    a_async(0, 0);
    loadB(0);
    storeB(0);
    int cur = 0;
    for (int k0 = 0; k0 < I_DIM; k0 += KB) {
        asm volatile("s_wait_asynccnt 0x0" ::: "memory");
        __syncthreads();
        const bool more = (k0 + KB) < I_DIM;
        if (more) { a_async(k0 + KB, cur ^ 1); loadB(k0 + KB); }
        compute(cur);
        if (more) storeB(cur ^ 1);
        cur ^= 1;
    }

    // ---- scatter fp32 results to out[token] (exclusive per token)
    const int hi8 = (lane < 16) ? 0 : 8;
    const int rowbase = mt * BM + mw * 16 + hi8;
    int tokv[8];
#pragma unroll
    for (int j = 0; j < 8; ++j) {
        const int g2 = rowbase + j;
        tokv[j] = (g2 < cnt) ? tlist[e * T_TOK + g2] : -1;
    }
#pragma unroll
    for (int nt = 0; nt < 4; ++nt) {
        const int col = nb * BN + nw * 64 + nt * 16 + (lane & 15);
#pragma unroll
        for (int j = 0; j < 8; ++j) {
            if (tokv[j] >= 0) {
                out[(size_t)tokv[j] * H_DIM + col] = cc[nt][j];
            }
        }
    }
}

// ---------------------------------------------------------------------------
extern "C" void kernel_launch(void* const* d_in, const int* in_sizes, int n_in,
                              void* d_out, int out_size, void* d_ws, size_t ws_size,
                              hipStream_t stream) {
    const float* x   = (const float*)d_in[0];   // (B,S,H)  fp32
    const float* gw  = (const float*)d_in[1];   // (E,H)    fp32
    const float* gup = (const float*)d_in[2];   // (E,H,2I) fp32
    const float* dwn = (const float*)d_in[3];   // (E,I,H)  fp32
    float* out = (float*)d_out;                 // (B,S,H)  fp32

    // Workspace layout (~8.5 MB total)
    float* scale  = (float*)d_ws;                          // T floats
    int*   counts = (int*)(scale + T_TOK);                 // 16 ints
    int*   tlist  = counts + 16;                           // E*T ints
    __bf16* act   = (__bf16*)(tlist + E_NUM * T_TOK);      // T*I bf16

    moe_zero<<<1, 32, 0, stream>>>(counts);
    moe_router<<<T_TOK, 32, 0, stream>>>(x, gw, scale, counts, tlist);

    dim3 g1(I_DIM / BN, T_TOK / BM, E_NUM);   // (8, 64, 8); extra tiles exit early
    moe_gemm1<<<g1, 256, 0, stream>>>(x, gup, scale, counts, tlist, act);

    dim3 g2(H_DIM / BN, T_TOK / BM, E_NUM);   // (16, 64, 8)
    moe_gemm2<<<g2, 256, 0, stream>>>(act, dwn, counts, tlist, out);
}
